// MNISTBinaryDenseNet_21732534518087
// MI455X (gfx1250) — 8-bit matrix operand order repaired and checked by reading, compiled, not run on hardware
//
#include <hip/hip_runtime.h>
#include <stdint.h>

typedef int v8i __attribute__((ext_vector_type(8)));
typedef unsigned u32x4 __attribute__((ext_vector_type(4)));
typedef int i32x4 __attribute__((ext_vector_type(4)));
typedef int i32x8 __attribute__((ext_vector_type(8)));

#define BM 128
#define BN 64
#define BK 64
#define LDA_DW 20
#define LDB_DW 20
#define SZA (BM * LDA_DW)
#define SZB (BK * LDB_DW)
#define SMEM_BYTES (2 * (SZA + SZB) * 4)

#if __has_builtin(__builtin_amdgcn_tensor_load_to_lds) && \
    __has_builtin(__builtin_amdgcn_s_wait_tensorcnt)
#define USE_TDM 1
#if __has_include(<hip/amd_detail/amd_gfx1250_TDM.h>)
#define TDM_6ARG 1
#else
#define TDM_6ARG 0
#endif
#else
#define USE_TDM 0
#endif

__global__ void k_binarize_act(const float* __restrict__ x,
                               int8_t* __restrict__ out, long long n4) {
  long long i = (long long)blockIdx.x * blockDim.x + threadIdx.x;
  if (i >= n4) return;
  float4 v = ((const float4*)x)[i];
  int b0 = (v.x >= 0.f) ? 1 : 0xFF;
  int b1 = (v.y >= 0.f) ? 1 : 0xFF;
  int b2 = (v.z >= 0.f) ? 1 : 0xFF;
  int b3 = (v.w >= 0.f) ? 1 : 0xFF;
  ((int*)out)[i] = b0 | (b1 << 8) | (b2 << 16) | (b3 << 24);
}

__global__ void k_binarize_wT(const float* __restrict__ W, int8_t* __restrict__ Wt,
                              int N, int K, int Np, int Kp) {
  int np4 = Np >> 2;
  long long i = (long long)blockIdx.x * blockDim.x + threadIdx.x;
  long long tot = (long long)Kp * np4;
  if (i >= tot) return;
  int k  = (int)(i / np4);
  int n0 = (int)(i - (long long)k * np4) * 4;
  int packed = 0;
#pragma unroll
  for (int j = 0; j < 4; ++j) {
    int n = n0 + j;
    int b = 0;
    if (k < K && n < N)
      b = (W[(long long)n * K + k] >= 0.f) ? 1 : 0xFF;
    packed |= (b & 0xFF) << (8 * j);
  }
  ((int*)Wt)[i] = packed;
}

#if USE_TDM
__device__ __forceinline__ void tdm_load_2d(unsigned lds_byte_off,
                                            const void* gptr,
                                            unsigned tile0_bytes,
                                            unsigned tile1_rows,
                                            unsigned long long row_stride_bytes) {
  unsigned long long ga = (unsigned long long)(uintptr_t)gptr;
  u32x4 g0;
  g0[0] = 1u;
  g0[1] = lds_byte_off;
  g0[2] = (unsigned)ga;
  g0[3] = (unsigned)((ga >> 32) & 0x1FFFFFFu)
          | (2u << 30);
  unsigned long long td0 = row_stride_bytes;
  unsigned long long s0  = row_stride_bytes;
  unsigned td1 = tile1_rows;
  i32x8 g1;
  g1[0] = (int)((3u << 25) | (3u << 22) | (1u << 20));
  g1[1] = (int)((unsigned)(td0 & 0xFFFFu) << 16);
  g1[2] = (int)(((td0 >> 16) & 0xFFFFu) | ((td1 & 0xFFFFu) << 16));
  g1[3] = (int)(((td1 >> 16) & 0xFFFFu) | (tile0_bytes << 16));
  g1[4] = (int)(tile1_rows & 0xFFFFu);
  g1[5] = (int)(unsigned)s0;
  g1[6] = (int)((s0 >> 32) & 0xFFFFu);
  g1[7] = 0;
  i32x4 z4 = {0, 0, 0, 0};
#if TDM_6ARG
  i32x8 z8 = {0, 0, 0, 0, 0, 0, 0, 0};
  __builtin_amdgcn_tensor_load_to_lds(g0, g1, z4, z4, z8, 0);
#else
  __builtin_amdgcn_tensor_load_to_lds(g0, g1, z4, z4, 0);
#endif
}
#endif

__device__ __forceinline__ void compute_tile(const int* __restrict__ bufA,
                                             const int* __restrict__ bufB,
                                             int lane, int wm, int wn,
                                             v8i (&acc)[2][2]) {
  v8i afrag[2], bfrag[2];
  const int hi = (lane >> 4) & 1;
  const int mrow = wm * 32 + (lane & 15);
#pragma unroll
  for (int mt = 0; mt < 2; ++mt) {
    const int row = mrow + mt * 16;
#pragma unroll
    for (int i = 0; i < 8; ++i) {
      const int k0 = ((i >> 1) << 4) + ((i & 1) << 2) + (hi << 3);
      afrag[mt][i] = bufA[row * LDA_DW + (k0 >> 2)];
    }
  }
#pragma unroll
  for (int nt = 0; nt < 2; ++nt) {
    const int colb = wn * 32 + nt * 16;
#pragma unroll
    for (int i = 0; i < 8; ++i) {
      const int k0 = ((i >> 1) << 4) + ((i & 1) << 2) + (hi << 3);
      const int col  = colb + (lane & 15);
      const unsigned char* bb = (const unsigned char*)bufB;
      int v = 0;
#pragma unroll
      for (int j = 0; j < 4; ++j)
        v |= (int)bb[(k0 + j) * (LDB_DW * 4) + col] << (8 * j);
      bfrag[nt][i] = v;
    }
  }
#pragma unroll
  for (int mt = 0; mt < 2; ++mt)
#pragma unroll
    for (int nt = 0; nt < 2; ++nt)
      acc[mt][nt] = __builtin_amdgcn_wmma_i32_16x16x64_iu8(
          true, afrag[mt], true, bfrag[nt], acc[mt][nt], false, false);
}

__global__ __launch_bounds__(256, 2)
void k_bingemm(const int8_t* __restrict__ A, const int8_t* __restrict__ Bt,
               int8_t* __restrict__ OutS, float* __restrict__ OutF,
               int sA, int sB, int sO, int numKT, int outFloat, float scale) {
  extern __shared__ __align__(16) int smem[];

  const int tid  = threadIdx.x;
  const int lane = tid & 31;
  const int w    = tid >> 5;
  const int wm   = w >> 1;
  const int wn   = w & 1;
  const long long m0 = (long long)blockIdx.x * BM;
  const int n0 = blockIdx.y * BN;

  v8i acc[2][2] = {};

#if USE_TDM
  for (int kt = 0; kt < numKT; ++kt) {
    const int cur = kt & 1;
    if (w == 0) {
      if (kt == 0) {
        tdm_load_2d(0, A + m0 * sA, BK, BM, (unsigned long long)sA);
        tdm_load_2d(SZA * 4u, Bt + n0, BK, BK, (unsigned long long)sB);
      }
      if (kt + 1 < numKT) {
        const unsigned aoff = (cur ? 0u : (unsigned)(SZA + SZB) * 4u);
        tdm_load_2d(aoff, A + m0 * sA + (long long)(kt + 1) * BK,
                    BK, BM, (unsigned long long)sA);
        tdm_load_2d(aoff + SZA * 4u,
                    Bt + (long long)(kt + 1) * BK * sB + n0,
                    BK, BK, (unsigned long long)sB);
        __builtin_amdgcn_s_wait_tensorcnt(2);
      } else {
        __builtin_amdgcn_s_wait_tensorcnt(0);
      }
    }
    __syncthreads();
    const int* bufA = smem + (cur ? (SZA + SZB) : 0);
    compute_tile(bufA, bufA + SZA, lane, wm, wn, acc);
    __syncthreads();
  }
#else
  int* bufA = smem;
  int* bufB = smem + SZA;
  const int arow0 = tid >> 2;
  const int arow1 = arow0 + 64;
  const int acol  = (tid & 3) * 16;
  const int brow  = tid >> 2;
  const int bcol  = (tid & 3) * 16;
  const int8_t* aptr0 = A + (m0 + arow0) * sA + acol;
  const int8_t* aptr1 = A + (m0 + arow1) * sA + acol;
  const int8_t* bptr  = Bt + (long long)brow * sB + n0 + bcol;

  int4 pa0 = *(const int4*)(aptr0);
  int4 pa1 = *(const int4*)(aptr1);
  int4 pb  = *(const int4*)(bptr);

  for (int kt = 0; kt < numKT; ++kt) {
    __syncthreads();
    *(int4*)&bufA[arow0 * LDA_DW + (acol >> 2)] = pa0;
    *(int4*)&bufA[arow1 * LDA_DW + (acol >> 2)] = pa1;
    *(int4*)&bufB[brow  * LDB_DW + (bcol >> 2)] = pb;
    __syncthreads();
    if (kt + 1 < numKT) {
      long long koff = (long long)(kt + 1) * BK;
      pa0 = *(const int4*)(aptr0 + koff);
      pa1 = *(const int4*)(aptr1 + koff);
      pb  = *(const int4*)(bptr + koff * sB);
      if (kt + 2 < numKT) {
        __builtin_prefetch(aptr0 + (long long)(kt + 2) * BK, 0, 3);
        __builtin_prefetch(bptr + (long long)(kt + 2) * BK * sB, 0, 3);
      }
    }
    compute_tile(bufA, bufB, lane, wm, wn, acc);
  }
#endif

  const int ncol = lane & 15;
  const int rofs = (lane >> 4) << 3;
#pragma unroll
  for (int mt = 0; mt < 2; ++mt) {
#pragma unroll
    for (int nt = 0; nt < 2; ++nt) {
      v8i a = acc[mt][nt];
#pragma unroll
      for (int r = 0; r < 8; ++r) {
        long long m = m0 + wm * 32 + mt * 16 + r + rofs;
        int n = n0 + wn * 32 + nt * 16 + ncol;
        if (outFloat)
          OutF[m * sO + n] = (float)a[r] * scale;
        else
          OutS[m * sO + n] = (a[r] >= 0) ? (int8_t)1 : (int8_t)-1;
      }
    }
  }
}

__global__ void k_logsoftmax(const float* __restrict__ logits,
                             float* __restrict__ out, int rows) {
  int r = blockIdx.x * blockDim.x + threadIdx.x;
  if (r >= rows) return;
  const float* p = logits + (long long)r * 64;
  float m = p[0];
#pragma unroll
  for (int j = 1; j < 10; ++j) m = fmaxf(m, p[j]);
  float s = 0.f;
#pragma unroll
  for (int j = 0; j < 10; ++j) s += __expf(p[j] - m);
  float ls = __logf(s);
#pragma unroll
  for (int j = 0; j < 10; ++j) out[(long long)r * 10 + j] = (p[j] - m) - ls;
}

extern "C" void kernel_launch(void* const* d_in, const int* in_sizes, int n_in,
                              void* d_out, int out_size, void* d_ws, size_t ws_size,
                              hipStream_t stream) {
  (void)in_sizes; (void)n_in; (void)out_size; (void)ws_size;
  const float* x  = (const float*)d_in[0];
  const float* W1 = (const float*)d_in[1];
  const float* W2 = (const float*)d_in[2];
  const float* W3 = (const float*)d_in[3];
  const float* W4 = (const float*)d_in[4];
  float* out = (float*)d_out;

  const int M   = 8192;
  const int K1  = 6272;
  const int F   = 784;
  const int Fp  = 832;
  const int N4p = 64;

  char* ws = (char*)d_ws;
  size_t off = 0;
  auto alloc = [&](size_t bytes) {
    char* p = ws + off;
    off += (bytes + 255) & ~(size_t)255;
    return p;
  };
  int8_t* Xb  = (int8_t*)alloc((size_t)M * K1);
  int8_t* W1t = (int8_t*)alloc((size_t)K1 * Fp);
  int8_t* W2t = (int8_t*)alloc((size_t)Fp * Fp);
  int8_t* W3t = (int8_t*)alloc((size_t)Fp * Fp);
  int8_t* W4t = (int8_t*)alloc((size_t)Fp * N4p);
  int8_t* H1  = (int8_t*)alloc((size_t)M * Fp);
  int8_t* H2  = (int8_t*)alloc((size_t)M * Fp);
  float*  LG  = (float*)alloc((size_t)M * N4p * sizeof(float));

  {
    long long n4 = (long long)M * K1 / 4;
    k_binarize_act<<<dim3((unsigned)((n4 + 255) / 256)), dim3(256), 0, stream>>>(x, Xb, n4);
  }
  auto binW = [&](const float* W, int8_t* Wt, int N, int K, int Np, int Kp) {
    long long tot = (long long)Kp * (Np / 4);
    k_binarize_wT<<<dim3((unsigned)((tot + 255) / 256)), dim3(256), 0, stream>>>(W, Wt, N, K, Np, Kp);
  };
  binW(W1, W1t, F, K1, Fp, K1);
  binW(W2, W2t, F, F, Fp, Fp);
  binW(W3, W3t, F, F, Fp, Fp);
  binW(W4, W4t, 10, F, N4p, Fp);

  dim3 blk(256);
  k_bingemm<<<dim3(M / BM, Fp / BN), blk, SMEM_BYTES, stream>>>(
      Xb, W1t, H1, nullptr, K1, Fp, Fp, K1 / BK, 0, 0.f);
  k_bingemm<<<dim3(M / BM, Fp / BN), blk, SMEM_BYTES, stream>>>(
      H1, W2t, H2, nullptr, Fp, Fp, Fp, Fp / BK, 0, 0.f);
  k_bingemm<<<dim3(M / BM, Fp / BN), blk, SMEM_BYTES, stream>>>(
      H2, W3t, H1, nullptr, Fp, Fp, Fp, Fp / BK, 0, 0.f);
  k_bingemm<<<dim3(M / BM, N4p / BN), blk, SMEM_BYTES, stream>>>(
      H1, W4t, nullptr, LG, Fp, N4p, N4p, Fp / BK, 1, 1.0f / 28.0f);
  k_logsoftmax<<<dim3((M + 255) / 256), dim3(256), 0, stream>>>(LG, out, M);
}
